// FuzzyMealyAnnealing_13022340842099
// MI455X (gfx1250) — matrix operand order repaired and checked by reading, not compiled, not run on hardware
//
#include <hip/hip_runtime.h>

// ---------------- types / helpers ----------------
typedef __attribute__((ext_vector_type(8)))  __bf16 v8bf;
typedef __attribute__((ext_vector_type(16))) __bf16 v16bf;
typedef __attribute__((ext_vector_type(8)))  float  v8f;

static __device__ inline unsigned short f2bf16u(float f) {
  unsigned u = __builtin_bit_cast(unsigned, f);
  u += 0x7fffu + ((u >> 16) & 1u);          // round-to-nearest-even
  return (unsigned short)(u >> 16);
}
static __device__ inline float bfu(unsigned short h) {
  return __builtin_bit_cast(float, (unsigned)h << 16);
}
static __device__ inline float bflo(unsigned u) { return __builtin_bit_cast(float, u << 16); }
static __device__ inline float bfhi(unsigned u) { return __builtin_bit_cast(float, u & 0xffff0000u); }

// CDNA5 async global->LDS copy (ASYNCcnt-tracked), 16B per lane-issue.
static __device__ inline void async_copy_b128(unsigned lds_addr, const void* gptr) {
#if defined(__HIP_DEVICE_COMPILE__)
  asm volatile("global_load_async_to_lds_b128 %0, %1, off"
               :: "v"(lds_addr), "v"(gptr)
               : "memory");
#endif
}
static __device__ inline void wait_async_zero() {
#if defined(__HIP_DEVICE_COMPILE__)
  asm volatile("s_wait_asynccnt 0x0" ::: "memory");
#endif
}

#define NS   512   // states
#define NI   128   // input symbols
#define NO   512   // outputs
#define NL   4096  // sequence length
#define NB   32    // chunk length
#define NC   128   // number of chunks  (NC*NB == NL)

// ---------------- 1) softmax row stats (max, sum) ----------------
__global__ __launch_bounds__(256) void softmax_stats_kernel(
    const float* __restrict__ src, float2* __restrict__ stats) {
  int wid  = (blockIdx.x * blockDim.x + threadIdx.x) >> 5;
  int lane = threadIdx.x & 31;
  if (wid >= NS * NI) return;
  const float* row = src + (size_t)wid * NS;
  float mx = -3.4e38f;
  for (int e = lane; e < NS; e += 32) mx = fmaxf(mx, row[e]);
  for (int s = 16; s; s >>= 1) mx = fmaxf(mx, __shfl_xor(mx, s, 32));
  float sum = 0.f;
  for (int e = lane; e < NS; e += 32) sum += __expf(row[e] - mx);
  for (int s = 16; s; s >>= 1) sum += __shfl_xor(sum, s, 32);
  if (lane == 0) stats[wid] = make_float2(mx, sum);
}

// ---------------- 2) normalize + transpose to bf16 ----------------
// src [k=512][i=128][j=512] f32  ->  dst [i=128][j=512][k=512] bf16
__global__ __launch_bounds__(256) void transpose_norm_kernel(
    const float* __restrict__ src, const float2* __restrict__ stats,
    unsigned short* __restrict__ dst) {
  __shared__ float tile[32][33];
  int i  = blockIdx.z;
  int kb = blockIdx.x * 32;
  int jb = blockIdx.y * 32;
#pragma unroll
  for (int rr = 0; rr < 4; rr++) {
    int r = threadIdx.y * 4 + rr;           // k within tile
    int c = threadIdx.x;                    // j within tile
    int k = kb + r, j = jb + c;
    float2 st = stats[k * NI + i];
    float  v  = src[((size_t)k * NI + i) * NS + j];
    tile[r][c] = __expf(v - st.x) / st.y;
  }
  __syncthreads();
#pragma unroll
  for (int rr = 0; rr < 4; rr++) {
    int r = threadIdx.y * 4 + rr;           // j within tile
    int c = threadIdx.x;                    // k within tile
    dst[((size_t)i * NS + (jb + r)) * NS + (kb + c)] = f2bf16u(tile[c][r]);
  }
}

// ---------------- 3) per-chunk product matrices (WMMA, LDS-staged B) ----------------
// P_c = T[x_{cB}] * T[x_{cB+1}] * ... * T[x_{cB+B-1}]   (bf16, f32 accum)
// A (current P) row-major [m][k] from global; B n-panels (64 cols x 512 k, 64 KB)
// async-staged into LDS (row pitch padded to 1040 B for conflict-free ds_load_b128).
#define PROW 520  // LDS panel row pitch in ushorts (512 + 8 pad)
__global__ __launch_bounds__(256) void chunk_product_kernel(
    const float* __restrict__ T, const float2* __restrict__ Tstats,
    const unsigned short* __restrict__ Tt, const int* __restrict__ seq,
    unsigned short* __restrict__ Pbuf) {
  __shared__ unsigned short panel[64 * PROW];   // ~66.5 KB
  int c = blockIdx.x;
  unsigned short* buf0 = Pbuf + (size_t)c * 2 * NS * NS;
  unsigned short* buf1 = buf0 + (size_t)NS * NS;
  int tid = threadIdx.x;

  // step 0: materialize normalized slice of first symbol into buf0 (row-major [m][k])
  int s0 = seq[c * NB];
  for (int idx = tid * 4; idx < NS * NS; idx += 256 * 4) {
    int m = idx >> 9, k = idx & (NS - 1);
    float2 st = Tstats[m * NI + s0];
    float4 v  = *(const float4*)(T + ((size_t)m * NI + s0) * NS + k);
    buf0[idx + 0] = f2bf16u(__expf(v.x - st.x) / st.y);
    buf0[idx + 1] = f2bf16u(__expf(v.y - st.x) / st.y);
    buf0[idx + 2] = f2bf16u(__expf(v.z - st.x) / st.y);
    buf0[idx + 3] = f2bf16u(__expf(v.w - st.x) / st.y);
  }
  __threadfence();
  __syncthreads();

  int lane = tid & 31, wave = tid >> 5;
  int half = lane >> 4, l16 = lane & 15;
  unsigned panel_base = (unsigned)(size_t)&panel[0];

  for (int r = 1; r < NB; r++) {
    const unsigned short* A = (r & 1) ? buf0 : buf1;
    unsigned short*       D = (r & 1) ? buf1 : buf0;
    const unsigned short* Bsl = Tt + (size_t)seq[c * NB + r] * NS * NS;

    for (int pass = 0; pass < 4; pass++) {
      int mt   = pass * 8 + wave;             // m-tile owned by this wave
      int rowm = mt * 16 + l16;
      // resident A strip: 16 k-steps of the 16x32 A tile for this lane
      v16bf areg[16];
#pragma unroll
      for (int ks = 0; ks < 16; ks++) {
        v8bf alo = *(const v8bf*)(A + (size_t)rowm * NS + ks * 32 + 8 * half);
        v8bf ahi = *(const v8bf*)(A + (size_t)rowm * NS + ks * 32 + 16 + 8 * half);
#pragma unroll
        for (int e = 0; e < 8; e++) { areg[ks][e] = alo[e]; areg[ks][8 + e] = ahi[e]; }
      }
      for (int pg = 0; pg < 8; pg++) {        // eight 64-column B panels
        __syncthreads();                      // everyone done reading previous panel
        const unsigned short* src = Bsl + (size_t)pg * 64 * NS;  // 64 contiguous rows
        for (int tr = tid; tr < 4096; tr += 256) {               // 4096 x 16B transfers
          int prow = tr >> 6, c16 = tr & 63;
          async_copy_b128(panel_base + (unsigned)(prow * PROW + c16 * 8) * 2,
                          src + (size_t)prow * NS + c16 * 8);
        }
        wait_async_zero();
        __syncthreads();                      // panel visible to all waves
#pragma unroll
        for (int ntl = 0; ntl < 4; ntl++) {
          int nl = ntl * 16 + l16;            // column row inside panel
          v8f acc = {0.f, 0.f, 0.f, 0.f, 0.f, 0.f, 0.f, 0.f};
#pragma unroll
          for (int ks = 0; ks < 16; ks++) {
            v8bf blo = *(const v8bf*)&panel[nl * PROW + ks * 32 + 8 * half];
            v8bf bhi = *(const v8bf*)&panel[nl * PROW + ks * 32 + 16 + 8 * half];
            v16bf b;
#pragma unroll
            for (int e = 0; e < 8; e++) { b[e] = blo[e]; b[8 + e] = bhi[e]; }
#if defined(__HIP_DEVICE_COMPILE__)
            acc = __builtin_amdgcn_wmma_f32_16x16x32_bf16(
                false, areg[ks], false, b, (short)0, acc, false, false);
#endif
          }
          int nt = pg * 4 + ntl;
#pragma unroll
          for (int rr = 0; rr < 8; rr++) {
            int row = mt * 16 + 8 * half + rr;
            D[(size_t)row * NS + nt * 16 + l16] = f2bf16u(acc[rr]);
          }
        }
      }
    }
    __threadfence();
    __syncthreads();
  }
}

// ---------------- 4) chain boundary states through P_c ----------------
__global__ __launch_bounds__(512) void boundary_kernel(
    const unsigned short* __restrict__ Pbuf, const int* __restrict__ init,
    float* __restrict__ bstates) {
  __shared__ float s[NS];
  __shared__ unsigned short pk[8][NS];
  int j = threadIdx.x;
  s[j] = (j == init[0]) ? 1.0f : 0.0f;
  bstates[j] = s[j];
  __syncthreads();
  for (int c = 0; c < NC; c++) {
    const unsigned short* P = Pbuf + ((size_t)2 * c + 1) * NS * NS;  // buf1 = final product
    float acc = 0.f;
    for (int kb = 0; kb < NS / 8; kb++) {
      int kr = j >> 6, g = j & 63;  // stage 8 rows cooperatively, vectorized
      *(uint4*)&pk[kr][g * 8] = *(const uint4*)(P + (size_t)(kb * 8 + kr) * NS + g * 8);
      __syncthreads();
#pragma unroll
      for (int e = 0; e < 8; e++) acc += s[kb * 8 + e] * bfu(pk[e][j]);
      __syncthreads();
    }
    s[j] = acc;
    bstates[(size_t)(c + 1) * NS + j] = acc;
    __syncthreads();
  }
}

// ---------------- 5) reconstruct interior states per chunk ----------------
__global__ __launch_bounds__(512) void chunk_states_kernel(
    const unsigned short* __restrict__ Tt, const int* __restrict__ seq,
    const float* __restrict__ bstates, float* __restrict__ states) {
  int c = blockIdx.x, j = threadIdx.x;
  __shared__ float s[NS];
  s[j] = bstates[(size_t)c * NS + j];
  states[(size_t)(c * NB) * NS + j] = s[j];
  __syncthreads();
  for (int r = 0; r < NB - 1; r++) {
    int sym = seq[c * NB + r];
    const unsigned short* row = Tt + ((size_t)sym * NS + j) * NS;  // s'[j] = sum_k s[k]*Tt[sym][j][k]
    float acc = 0.f;
#pragma unroll 2
    for (int k = 0; k < NS; k += 8) {
      uint4 v = *(const uint4*)(row + k);
      acc += s[k + 0] * bflo(v.x) + s[k + 1] * bfhi(v.x);
      acc += s[k + 2] * bflo(v.y) + s[k + 3] * bfhi(v.y);
      acc += s[k + 4] * bflo(v.z) + s[k + 5] * bfhi(v.z);
      acc += s[k + 6] * bflo(v.w) + s[k + 7] * bfhi(v.w);
    }
    __syncthreads();
    s[j] = acc;
    states[(size_t)(c * NB + r + 1) * NS + j] = acc;
    __syncthreads();
  }
}

// ---------------- 6) group timesteps by symbol ----------------
__global__ void group_zero_kernel(int* counts, int* cursor) {
  int t = threadIdx.x;
  if (t < NI) { counts[t] = 0; cursor[t] = 0; }
}
__global__ void group_count_kernel(const int* __restrict__ seq, int* counts) {
  int t = blockIdx.x * blockDim.x + threadIdx.x;
  if (t < NL) atomicAdd(&counts[seq[t]], 1);
}
__global__ void group_prefix_kernel(const int* __restrict__ counts, int* offs) {
  if (threadIdx.x == 0) {
    int a = 0;
    for (int i = 0; i < NI; i++) { offs[i] = a; a += counts[i]; }
    offs[NI] = a;
  }
}
__global__ void group_scatter_kernel(const int* __restrict__ seq, const int* __restrict__ offs,
                                     int* cursor, int* list) {
  int t = blockIdx.x * blockDim.x + threadIdx.x;
  if (t < NL) {
    int sym = seq[t];
    int p = atomicAdd(&cursor[sym], 1);
    list[offs[sym] + p] = t;
  }
}

// ---------------- 7) outputs: per-symbol GEMM via WMMA ----------------
// out[t, :] = states[t] @ O_soft[:, x_t, :], batched over all t with x_t == i.
__global__ __launch_bounds__(256) void output_gemm_kernel(
    const float* __restrict__ states, const unsigned short* __restrict__ Ot,
    const int* __restrict__ offs, const int* __restrict__ counts,
    const int* __restrict__ list, float* __restrict__ out) {
  int nt = blockIdx.x;                 // 0..31 output n-tile (16 cols)
  int i  = blockIdx.y;                 // symbol
  int Mi = counts[i];
  if (Mi == 0) return;
  int off  = offs[i];
  int lane = threadIdx.x & 31, wave = threadIdx.x >> 5;
  int half = lane >> 4, l16 = lane & 15;
  const unsigned short* Bsl = Ot + (size_t)i * NO * NS;  // [o][k] contiguous k
  int nmt = (Mi + 15) >> 4;
  for (int mt = wave; mt < nmt; mt += 8) {
    int gr  = mt * 16 + l16;
    int t_l = list[off + (gr < Mi ? gr : 0)];
    const float* arow = states + (size_t)t_l * NS;
    int coln = nt * 16 + l16;
    v8f acc = {0.f, 0.f, 0.f, 0.f, 0.f, 0.f, 0.f, 0.f};
#pragma unroll
    for (int ks = 0; ks < 16; ks++) {
      float4 f0 = *(const float4*)(arow + ks * 32 + 8 * half);
      float4 f1 = *(const float4*)(arow + ks * 32 + 8 * half + 4);
      float4 f2 = *(const float4*)(arow + ks * 32 + 16 + 8 * half);
      float4 f3 = *(const float4*)(arow + ks * 32 + 16 + 8 * half + 4);
      v16bf a;
      a[0]  = (__bf16)f0.x; a[1]  = (__bf16)f0.y; a[2]  = (__bf16)f0.z; a[3]  = (__bf16)f0.w;
      a[4]  = (__bf16)f1.x; a[5]  = (__bf16)f1.y; a[6]  = (__bf16)f1.z; a[7]  = (__bf16)f1.w;
      a[8]  = (__bf16)f2.x; a[9]  = (__bf16)f2.y; a[10] = (__bf16)f2.z; a[11] = (__bf16)f2.w;
      a[12] = (__bf16)f3.x; a[13] = (__bf16)f3.y; a[14] = (__bf16)f3.z; a[15] = (__bf16)f3.w;
      v8bf blo = *(const v8bf*)(Bsl + (size_t)coln * NS + ks * 32 + 8 * half);
      v8bf bhi = *(const v8bf*)(Bsl + (size_t)coln * NS + ks * 32 + 16 + 8 * half);
      v16bf b;
#pragma unroll
      for (int e = 0; e < 8; e++) { b[e] = blo[e]; b[8 + e] = bhi[e]; }
#if defined(__HIP_DEVICE_COMPILE__)
      acc = __builtin_amdgcn_wmma_f32_16x16x32_bf16(
          false, a, false, b, (short)0, acc, false, false);
#endif
    }
#pragma unroll
    for (int rr = 0; rr < 8; rr++) {
      int gm = mt * 16 + 8 * half + rr;
      if (gm < Mi) {
        int t = list[off + gm];
        out[(size_t)t * NO + nt * 16 + l16] = acc[rr];
      }
    }
  }
}

// ---------------- launch ----------------
extern "C" void kernel_launch(void* const* d_in, const int* in_sizes, int n_in,
                              void* d_out, int out_size, void* d_ws, size_t ws_size,
                              hipStream_t stream) {
  const float* Tp  = (const float*)d_in[0];
  const float* Op  = (const float*)d_in[1];
  const int*  seq  = (const int*)d_in[2];
  const int*  init = (const int*)d_in[3];
  float* out = (float*)d_out;

  char* base = (char*)d_ws;
  size_t off = 0;
  auto alloc = [&](size_t n) {
    char* p = base + off;
    off = (off + n + 255) & ~(size_t)255;
    return p;
  };
  unsigned short* Tt   = (unsigned short*)alloc((size_t)NI * NS * NS * 2);
  unsigned short* Ot   = (unsigned short*)alloc((size_t)NI * NO * NS * 2);
  float2*         Tst  = (float2*)alloc((size_t)NS * NI * sizeof(float2));
  float2*         Ost  = (float2*)alloc((size_t)NS * NI * sizeof(float2));
  unsigned short* Pbuf = (unsigned short*)alloc((size_t)NC * 2 * NS * NS * 2);
  float*          bst  = (float*)alloc((size_t)(NC + 1) * NS * 4);
  float*          st   = (float*)alloc((size_t)NL * NS * 4);
  int*            cnt  = (int*)alloc(NI * 4);
  int*            ofs  = (int*)alloc((NI + 1) * 4);
  int*            cur  = (int*)alloc(NI * 4);
  int*            lst  = (int*)alloc(NL * 4);

  int statBlocks = (NS * NI * 32 + 255) / 256;  // one wave per row
  softmax_stats_kernel<<<statBlocks, 256, 0, stream>>>(Tp, Tst);
  softmax_stats_kernel<<<statBlocks, 256, 0, stream>>>(Op, Ost);
  transpose_norm_kernel<<<dim3(16, 16, NI), dim3(32, 8), 0, stream>>>(Tp, Tst, Tt);
  transpose_norm_kernel<<<dim3(16, 16, NI), dim3(32, 8), 0, stream>>>(Op, Ost, Ot);
  chunk_product_kernel<<<NC, 256, 0, stream>>>(Tp, Tst, Tt, seq, Pbuf);
  boundary_kernel<<<1, NS, 0, stream>>>(Pbuf, init, bst);
  chunk_states_kernel<<<NC, NS, 0, stream>>>(Tt, seq, bst, st);
  group_zero_kernel<<<1, 128, 0, stream>>>(cnt, cur);
  group_count_kernel<<<NL / 256, 256, 0, stream>>>(seq, cnt);
  group_prefix_kernel<<<1, 32, 0, stream>>>(cnt, ofs);
  group_scatter_kernel<<<NL / 256, 256, 0, stream>>>(seq, ofs, cur, lst);
  output_gemm_kernel<<<dim3(NO / 16, NI), 256, 0, stream>>>(st, Ot, ofs, cnt, lst, out);
}
